// AttentionModule_49804440764736
// MI455X (gfx1250) — matrix operand order repaired and checked by reading, not compiled, not run on hardware
//
#include <hip/hip_runtime.h>
#include <hip/hip_bf16.h>
#include <cmath>

typedef __attribute__((ext_vector_type(16))) __bf16 v16bf;
typedef __attribute__((ext_vector_type(8)))  float  v8f;

static constexpr int BB = 8, NQ = 2048, NK = 2048, DD = 1024;
static constexpr int LDA_S = 40;    // padded LDS row stride, A tile (elems): 20 DW -> conflict-free-ish
static constexpr int LDB_S = 136;   // padded LDS row stride, B tile (elems): 68 DW -> lanes spread over banks

// ---------------------------------------------------------------------------
// CDNA5 async copy: 16 bytes/lane global -> LDS, tracked by ASYNCcnt
// ---------------------------------------------------------------------------
static __device__ inline void async_cp_b128(const void* g, void* l) {
    unsigned lds_off = (unsigned)(unsigned long long)(uintptr_t)l;   // low 32b of generic = LDS offset
    unsigned long long ga = (unsigned long long)(uintptr_t)g;
    asm volatile("global_load_async_to_lds_b128 %0, %1, off"
                 :: "v"(lds_off), "v"(ga) : "memory");
}
static __device__ inline void wait_async0() {
    asm volatile("s_wait_asynccnt 0" ::: "memory");
}

// ---------------------------------------------------------------------------
// fp32 -> bf16 conversion (vectorized by 4)
// ---------------------------------------------------------------------------
__global__ __launch_bounds__(256)
void cvt_f32_bf16(const float* __restrict__ s, __bf16* __restrict__ d, int n) {
    int i = (blockIdx.x * blockDim.x + threadIdx.x) * 4;
    if (i + 3 < n) {
        float4 f = *reinterpret_cast<const float4*>(s + i);
        union { __bf16 b[4]; ushort4 u; } o;
        o.b[0] = (__bf16)f.x; o.b[1] = (__bf16)f.y;
        o.b[2] = (__bf16)f.z; o.b[3] = (__bf16)f.w;
        *reinterpret_cast<ushort4*>(d + i) = o.u;
    }
}

// Load a v16bf WMMA operand from LDS as two 16-byte chunks
static __device__ inline v16bf ld_op16(const __bf16* p, int hi_off_elems) {
    union { v16bf v; float4 f[2]; } u;
    u.f[0] = *reinterpret_cast<const float4*>(p);
    u.f[1] = *reinterpret_cast<const float4*>(p + hi_off_elems);
    return u.v;
}

// ---------------------------------------------------------------------------
// Batched bf16 WMMA GEMM, double-buffered LDS, async global->LDS staging.
//   TRANSB==1 : C[M,N] = A[M,K] * B[N,K]^T      (both row-major, K contiguous)
//   TRANSB==0 : C[M,N] = A[M,K] * B[K,N]        (both row-major)
//   EPI==0    : store bf16  C = alpha*acc
//   EPI==1    : store fp32  C = alpha*acc + Res
// WG tile 128(M) x 128(N), BK=32, 256 threads = 8 waves (2 M x 4 N),
// wave tile 64x32 = 4x2 WMMA tiles = 8 v_wmma per K-step, 1 barrier per step.
// Grid: x = N/128, y = M/128, z = batch.
// ---------------------------------------------------------------------------
template <int TRANSB, int EPI>
__global__ __launch_bounds__(256)
void gemm_bf16_wmma(const __bf16* __restrict__ A, const __bf16* __restrict__ Bm,
                    void* __restrict__ Cout, const float* __restrict__ Res,
                    int K, int lda, int ldb, int ldc,
                    long long batA, long long batB, long long batC, long long batR,
                    float alpha) {
    __shared__ __bf16 lA[2][128 * LDA_S];   // [row][k], padded
    __shared__ __bf16 lB[2][32 * LDB_S];    // [k][n],   padded

    const int tid  = threadIdx.x;
    const int lane = tid & 31;
    const int wave = tid >> 5;
    const int wm   = (wave & 1) * 64;    // wave M offset within 128
    const int wn   = (wave >> 1) * 32;   // wave N offset within 128

    const long long tileM = (long long)blockIdx.y * 128;
    const long long tileN = (long long)blockIdx.x * 128;

    A  += (long long)blockIdx.z * batA;
    Bm += (long long)blockIdx.z * batB;

    // A fill: 128x32 tile, 2 chunks of 8 bf16 per thread
    const int aRow = tid >> 2;           // 0..63 (+64 for chunk 1)
    const int aCol = (tid & 3) * 8;

    v8f acc[4][2] = {};

    auto stageA = [&](int k0, int buf) {
#pragma unroll
        for (int c = 0; c < 2; ++c) {
            const int row = aRow + 64 * c;
            const __bf16* g = A + (tileM + row) * (long long)lda + (k0 + aCol);
            async_cp_b128(g, &lA[buf][row * LDA_S + aCol]);
        }
    };
    auto stageB_async = [&](int k0, int buf) {      // !TRANSB: rows are K, contiguous N
#pragma unroll
        for (int c = 0; c < 2; ++c) {
            const int lin = tid + 256 * c;
            const int kr = lin >> 4, nc = (lin & 15) * 8;
            const __bf16* g = Bm + (long long)(k0 + kr) * ldb + (tileN + nc);
            async_cp_b128(g, &lB[buf][kr * LDB_S + nc]);
        }
    };
    auto loadB = [&](int k0, float4* r) {           // TRANSB: rows are N, contiguous K
#pragma unroll
        for (int c = 0; c < 2; ++c) {
            const int lin = tid + 256 * c;
            const int n = lin >> 2, kc = (lin & 3) * 8;
            r[c] = *reinterpret_cast<const float4*>(
                Bm + (tileN + n) * (long long)ldb + (k0 + kc));
        }
    };
    auto scatterB = [&](int buf, const float4* r) { // transpose into [k][n]
#pragma unroll
        for (int c = 0; c < 2; ++c) {
            const int lin = tid + 256 * c;
            const int n = lin >> 2, kc = (lin & 3) * 8;
            union { float4 f; __bf16 b[8]; } u; u.f = r[c];
#pragma unroll
            for (int j = 0; j < 8; ++j) lB[buf][(kc + j) * LDB_S + n] = u.b[j];
        }
    };

    // WMMA operand lane mapping (CDNA5 ISA layout tables)
    const int ar = lane & 15;            // A: row within 16
    const int ak = (lane >> 4) * 8;      // A: K base (0/8), +16 for hi half
    auto compute = [&](int buf) {
        v16bf av[4], bv[2];
#pragma unroll
        for (int i = 0; i < 4; ++i)
            av[i] = ld_op16(&lA[buf][(wm + i * 16 + ar) * LDA_S + ak], 16);
#pragma unroll
        for (int j = 0; j < 2; ++j)
#pragma unroll
            for (int s = 0; s < 16; ++s)
                bv[j][s] = lB[buf][(ak + s + (s & 8)) * LDB_S + wn + j * 16 + ar];
#pragma unroll
        for (int i = 0; i < 4; ++i)
#pragma unroll
            for (int j = 0; j < 2; ++j)
                acc[i][j] = __builtin_amdgcn_wmma_f32_16x16x32_bf16(
                    false, av[i], false, bv[j], (short)0, acc[i][j], false, false);
    };

    // ---- prologue: stage k0 = 0 into buffer 0 ----
    {
        float4 br[2];
        stageA(0, 0);
        if (TRANSB) { loadB(0, br); scatterB(0, br); }
        else        stageB_async(0, 0);
        wait_async0();
        __syncthreads();
    }

    // ---- main loop: compute buf(cur) while staging buf(cur^1) ----
    int cur = 0;
    for (int k0 = 0; k0 < K; k0 += 32) {
        const int  nxt  = cur ^ 1;
        const bool more = (k0 + 32) < K;
        float4 brn[2];
        if (more) {
            stageA(k0 + 32, nxt);
            if (TRANSB) loadB(k0 + 32, brn);
            else        stageB_async(k0 + 32, nxt);
        }
        compute(cur);
        if (more) {
            if (TRANSB) scatterB(nxt, brn);
            wait_async0();
            __syncthreads();
        }
        cur = nxt;
    }

    // ---- epilogue ----
    const int cn = lane & 15;            // N within 16x16 tile
    const int cm = (lane >> 4) * 8;      // M offset (0 or 8)
#pragma unroll
    for (int i = 0; i < 4; ++i) {
#pragma unroll
        for (int j = 0; j < 2; ++j) {
#pragma unroll
            for (int r = 0; r < 8; ++r) {
                const long long m = tileM + wm + i * 16 + cm + r;
                const long long n = tileN + wn + j * 16 + cn;
                const float v = acc[i][j][r] * alpha;
                if (EPI == 0) {
                    __bf16* C = (__bf16*)Cout + (long long)blockIdx.z * batC;
                    C[m * ldc + n] = (__bf16)v;
                } else {
                    float* C = (float*)Cout + (long long)blockIdx.z * batC;
                    const float* R = Res + (long long)blockIdx.z * batR;
                    C[m * ldc + n] = v + R[m * ldc + n];
                }
            }
        }
    }
}

// ---------------------------------------------------------------------------
// In-place row softmax over bf16 scores (row = 2048 elems, 256 threads).
// ---------------------------------------------------------------------------
__global__ __launch_bounds__(256)
void softmax_rows_bf16(__bf16* __restrict__ S, int ncols) {
    const long long row = blockIdx.x;
    __bf16* p = S + row * (long long)ncols;
    const int tid = threadIdx.x, lane = tid & 31, wave = tid >> 5;

    float v[8];
    {
        union { float4 f; __bf16 b[8]; } u;
        u.f = *reinterpret_cast<const float4*>(p + tid * 8);
#pragma unroll
        for (int j = 0; j < 8; ++j) v[j] = (float)u.b[j];
    }

    __shared__ float red[8];

    float mx = -INFINITY;
#pragma unroll
    for (int j = 0; j < 8; ++j) mx = fmaxf(mx, v[j]);
    for (int o = 16; o > 0; o >>= 1) mx = fmaxf(mx, __shfl_xor(mx, o, 32));
    if (lane == 0) red[wave] = mx;
    __syncthreads();
    if (tid < 32) {
        float t = (lane < 8) ? red[lane] : -INFINITY;
        for (int o = 4; o > 0; o >>= 1) t = fmaxf(t, __shfl_xor(t, o, 32));
        if (lane == 0) red[0] = t;
    }
    __syncthreads();
    mx = red[0];
    __syncthreads();

    float e[8], s = 0.f;
#pragma unroll
    for (int j = 0; j < 8; ++j) { e[j] = __expf(v[j] - mx); s += e[j]; }
    for (int o = 16; o > 0; o >>= 1) s += __shfl_xor(s, o, 32);
    if (lane == 0) red[wave] = s;
    __syncthreads();
    if (tid < 32) {
        float t = (lane < 8) ? red[lane] : 0.f;
        for (int o = 4; o > 0; o >>= 1) t += __shfl_xor(t, o, 32);
        if (lane == 0) red[0] = t;
    }
    __syncthreads();
    const float inv = 1.0f / red[0];

    union { float4 f; __bf16 b[8]; } o;
#pragma unroll
    for (int j = 0; j < 8; ++j) o.b[j] = (__bf16)(e[j] * inv);
    *reinterpret_cast<float4*>(p + tid * 8) = o.f;
}

// ---------------------------------------------------------------------------
// Host launcher
// ---------------------------------------------------------------------------
extern "C" void kernel_launch(void* const* d_in, const int* in_sizes, int n_in,
                              void* d_out, int out_size, void* d_ws, size_t ws_size,
                              hipStream_t stream) {
    const float* qin = (const float*)d_in[0];   // (B, NQ, D)
    const float* kin = (const float*)d_in[1];   // (B, NK, D)
    const float* Wq  = (const float*)d_in[2];   // (D, D)
    const float* Wk  = (const float*)d_in[3];
    const float* Wv  = (const float*)d_in[4];
    float* out = (float*)d_out;                 // (B, NK, D)

    const long long nX = (long long)BB * NQ * DD;     // 16,777,216
    const long long nW = (long long)DD * DD;          //  1,048,576
    const long long nS = (long long)BB * NK * NQ;     // 33,554,432

    char* ws = (char*)d_ws;
    __bf16* Xq  = (__bf16*)ws;  ws += nX * 2;
    __bf16* Xk  = (__bf16*)ws;  ws += nX * 2;
    __bf16* Wqb = (__bf16*)ws;  ws += nW * 2;
    __bf16* Wkb = (__bf16*)ws;  ws += nW * 2;
    __bf16* Wvb = (__bf16*)ws;  ws += nW * 2;
    __bf16* qb  = (__bf16*)ws;  ws += nX * 2;
    __bf16* kb  = (__bf16*)ws;  ws += nX * 2;
    __bf16* vb  = (__bf16*)ws;  ws += nX * 2;
    __bf16* S   = (__bf16*)ws;  ws += nS * 2;         // total ~230 MB

    // Stage 1: fp32 -> bf16
    cvt_f32_bf16<<<(int)(nX / 4 / 256), 256, 0, stream>>>(qin, Xq, (int)nX);
    cvt_f32_bf16<<<(int)(nX / 4 / 256), 256, 0, stream>>>(kin, Xk, (int)nX);
    cvt_f32_bf16<<<(int)(nW / 4 / 256), 256, 0, stream>>>(Wq, Wqb, (int)nW);
    cvt_f32_bf16<<<(int)(nW / 4 / 256), 256, 0, stream>>>(Wk, Wkb, (int)nW);
    cvt_f32_bf16<<<(int)(nW / 4 / 256), 256, 0, stream>>>(Wv, Wvb, (int)nW);

    // Stage 2: projections  y = X @ W^T   (M = B*NQ = 16384, N = K = 1024)
    {
        dim3 g(DD / 128, (BB * NQ) / 128, 1);
        gemm_bf16_wmma<1, 0><<<g, 256, 0, stream>>>(Xq, Wqb, qb, nullptr,
            DD, DD, DD, DD, 0, 0, 0, 0, 1.0f);
        gemm_bf16_wmma<1, 0><<<g, 256, 0, stream>>>(Xk, Wkb, kb, nullptr,
            DD, DD, DD, DD, 0, 0, 0, 0, 1.0f);
        gemm_bf16_wmma<1, 0><<<g, 256, 0, stream>>>(Xq, Wvb, vb, nullptr,
            DD, DD, DD, DD, 0, 0, 0, 0, 1.0f);
    }

    // Stage 3: scores  S[b] = (k[b] @ q[b]^T) / sqrt(D)   (M=NK, N=NQ, K=D)
    {
        dim3 g(NQ / 128, NK / 128, BB);
        gemm_bf16_wmma<1, 0><<<g, 256, 0, stream>>>(kb, qb, S, nullptr,
            DD, DD, DD, NQ,
            (long long)NK * DD, (long long)NQ * DD, (long long)NK * NQ, 0,
            1.0f / 32.0f /* 1/sqrt(1024) */);
    }

    // Stage 4: softmax over NQ (last axis), in place, bf16 P
    softmax_rows_bf16<<<BB * NK, 256, 0, stream>>>(S, NQ);

    // Stage 5: context = P @ v + key_input   (M=NK, N=D, K=NQ)
    {
        dim3 g(DD / 128, NK / 128, BB);
        gemm_bf16_wmma<0, 1><<<g, 256, 0, stream>>>(S, vb, out, kin,
            NQ, NQ, DD, DD,
            (long long)NK * NQ, (long long)NQ * DD,
            (long long)NK * DD, (long long)NK * DD,
            1.0f);
    }
}
